// EncoderBlock_86895778333355
// MI455X (gfx1250) — matrix operand order repaired and checked by reading, not compiled, not run on hardware
//
#include <hip/hip_runtime.h>
#include <stdint.h>

// ---------------------------------------------------------------------------
// Types for CDNA5 WMMA (wave32): v_wmma_f32_16x16x32_bf16
// ---------------------------------------------------------------------------
typedef __bf16 bf16;
typedef __attribute__((ext_vector_type(16))) __bf16 v16bf;
typedef __attribute__((ext_vector_type(8)))  float  v8f;

#define WMMA_BF16(A, B, C) \
    __builtin_amdgcn_wmma_f32_16x16x32_bf16(false, (A), false, (B), (short)0, (C), false, false)

// ---------------------------------------------------------------------------
// Fragment helpers (A and B both use contiguous k order: K = 16*(lane>>4) + element).
// All loads are "uniform base + 32-bit lane offset" so the backend can use
// SGPR-base + VGPR-offset global addressing (1 VGPR per address, not 2).
// ---------------------------------------------------------------------------
__device__ __forceinline__ unsigned pack2(bf16 lo, bf16 hi) {
    unsigned short a = __builtin_bit_cast(unsigned short, lo);
    unsigned short b = __builtin_bit_cast(unsigned short, hi);
    return (unsigned)a | ((unsigned)b << 16);
}
__device__ __forceinline__ unsigned packf2(float lo, float hi) {
    return pack2((bf16)lo, (bf16)hi);
}

// A-matrix 16x32 bf16 fragment: lane&15 = row; per-dword K = 2*i + 16*(lane>>4)
// Dwords are contiguous in two 16B chunks -> merges into 2x global_load_b128.
__device__ __forceinline__ v16bf load_a_frag(const bf16* __restrict__ base, unsigned off,
                                             int khalf, unsigned mask) {
    union { v16bf v; unsigned u[8]; } f;
#pragma unroll
    for (int i = 0; i < 8; ++i) {
        int k = 2 * i + khalf * 16;
        f.u[i] = (*(const unsigned*)(base + off + k)) & mask;
    }
    return f.v;
}

// B-matrix 32x16 fragment, column-major storage (pair K=2v,2v+1 contiguous).
// off must address W[col][kk + 16*(lane>>4)]. 8 contiguous dwords -> 2x b128.
__device__ __forceinline__ v16bf load_b_ct(const bf16* __restrict__ base, unsigned off) {
    union { v16bf v; unsigned u[8]; } f;
#pragma unroll
    for (int i = 0; i < 8; ++i) f.u[i] = *(const unsigned*)(base + off + 2 * i);
    return f.v;
}

// 8 WMMAs for a 32x64 tile: 2 A-frags x 4 B-frags
__device__ __forceinline__ void wmma8(v8f acc[8], v16bf a0, v16bf a1,
                                      v16bf b0, v16bf b1, v16bf b2, v16bf b3) {
    acc[0] = WMMA_BF16(a0, b0, acc[0]);
    acc[1] = WMMA_BF16(a0, b1, acc[1]);
    acc[2] = WMMA_BF16(a0, b2, acc[2]);
    acc[3] = WMMA_BF16(a0, b3, acc[3]);
    acc[4] = WMMA_BF16(a1, b0, acc[4]);
    acc[5] = WMMA_BF16(a1, b1, acc[5]);
    acc[6] = WMMA_BF16(a1, b2, acc[6]);
    acc[7] = WMMA_BF16(a1, b3, acc[7]);
}

// ---------------------------------------------------------------------------
// Conversion / repack kernels
// ---------------------------------------------------------------------------
__global__ void cvt_f32_bf16_kernel(const float* __restrict__ in, bf16* __restrict__ out, size_t n) {
    size_t i = (size_t)blockIdx.x * blockDim.x + threadIdx.x;
    if (i < n) out[i] = (bf16)in[i];
}

// conv weight [O][C][3] f32 -> [tap][O][C] bf16
__global__ void repack_conv_kernel(const float* __restrict__ w, bf16* __restrict__ wr,
                                   int O, int C) {
    size_t i = (size_t)blockIdx.x * blockDim.x + threadIdx.x;
    size_t total = (size_t)O * C * 3;
    if (i >= total) return;
    int tap = (int)(i % 3);
    size_t oc = i / 3;
    int c = (int)(oc % C);
    int o = (int)(oc / C);
    wr[((size_t)tap * O + o) * C + c] = (bf16)w[i];
}

// V [B,S,H*DH] bf16 -> Vt [B,H,DH,S] bf16 (so P@V B-frags are contiguous-pair loads)
__global__ void transpose_v_kernel(const bf16* __restrict__ V, bf16* __restrict__ Vt,
                                   int S, int E, int H, int DH) {
    size_t i = (size_t)blockIdx.x * blockDim.x + threadIdx.x;
    size_t total = (size_t)S * E * 8;  // B*S*E with B=8
    if (i >= total) return;
    int j = (int)(i % S);
    size_t r = i / S;
    int d = (int)(r % DH);
    size_t r2 = r / DH;
    int h = (int)(r2 % H);
    int b = (int)(r2 / H);
    Vt[i] = V[((size_t)(b * S + j)) * E + h * DH + d];
}

// ---------------------------------------------------------------------------
// Generic GEMM: C[M,N] = A_bf16[M,K] @ W_bf16[N,K]^T + bias (+residual)
// One wave per 32x64 tile, register double-buffered K loop.
// grid = (N/64, M/32), 32 threads.
// ---------------------------------------------------------------------------
__global__ void __launch_bounds__(32, 1)
gemm_bias_kernel(const bf16* __restrict__ A, const bf16* __restrict__ W,
                 const float* __restrict__ bias, const float* __restrict__ residual,
                 bf16* __restrict__ outb, float* __restrict__ outf,
                 int M, int N, int K) {
    const int lane  = threadIdx.x & 31;
    const int lrow  = lane & 15;
    const int khalf = lane >> 4;
    const int mBase = blockIdx.y << 5;
    const int nBase = blockIdx.x << 6;

    v8f acc[8];
#pragma unroll
    for (int t = 0; t < 8; ++t)
#pragma unroll
        for (int r = 0; r < 8; ++r) acc[t][r] = 0.0f;

    const unsigned aoff0 = (unsigned)(mBase + lrow) * (unsigned)K;
    const unsigned aoff1 = aoff0 + 16u * (unsigned)K;
    const unsigned kh16  = (unsigned)(khalf * 16);
    const unsigned woff0 = (unsigned)(nBase +  0 + lrow) * (unsigned)K + kh16;
    const unsigned woff1 = (unsigned)(nBase + 16 + lrow) * (unsigned)K + kh16;
    const unsigned woff2 = (unsigned)(nBase + 32 + lrow) * (unsigned)K + kh16;
    const unsigned woff3 = (unsigned)(nBase + 48 + lrow) * (unsigned)K + kh16;

    v16bf a0 = load_a_frag(A, aoff0, khalf, 0xFFFFFFFFu);
    v16bf a1 = load_a_frag(A, aoff1, khalf, 0xFFFFFFFFu);
    v16bf b0 = load_b_ct(W, woff0), b1 = load_b_ct(W, woff1);
    v16bf b2 = load_b_ct(W, woff2), b3 = load_b_ct(W, woff3);

    for (unsigned kk = 32; kk < (unsigned)K; kk += 32) {
        v16bf a0n = load_a_frag(A, aoff0 + kk, khalf, 0xFFFFFFFFu);
        v16bf a1n = load_a_frag(A, aoff1 + kk, khalf, 0xFFFFFFFFu);
        v16bf b0n = load_b_ct(W, woff0 + kk), b1n = load_b_ct(W, woff1 + kk);
        v16bf b2n = load_b_ct(W, woff2 + kk), b3n = load_b_ct(W, woff3 + kk);
        wmma8(acc, a0, a1, b0, b1, b2, b3);
        a0 = a0n; a1 = a1n; b0 = b0n; b1 = b1n; b2 = b2n; b3 = b3n;
    }
    wmma8(acc, a0, a1, b0, b1, b2, b3);

#pragma unroll
    for (int half = 0; half < 2; ++half) {
#pragma unroll
        for (int t = 0; t < 4; ++t) {
            int col = nBase + t * 16 + lrow;
            float bvv = bias ? bias[col] : 0.0f;
#pragma unroll
            for (int r = 0; r < 8; ++r) {
                unsigned row = (unsigned)(mBase + half * 16 + r + khalf * 8);
                unsigned idx = row * (unsigned)N + (unsigned)col;
                float v = acc[half * 4 + t][r] + bvv;
                if (residual) v += residual[idx];
                if (outf) outf[idx] = v;
                if (outb) outb[idx] = (bf16)v;
            }
        }
    }
}

// ---------------------------------------------------------------------------
// Attention scores: scores[b,h,i,j] = (Q[b,i,h,:] . K[b,j,h,:]) / 8
// grid = (S/64, S/32, B*H), 32 threads, 32x64 tile per wave (K = DH = 64).
// ---------------------------------------------------------------------------
__global__ void __launch_bounds__(32, 1)
attn_scores_kernel(const bf16* __restrict__ Q, const bf16* __restrict__ Kmat,
                   float* __restrict__ scores, int S, int E, int H, int DH) {
    const int lane  = threadIdx.x & 31;
    const int lrow  = lane & 15;
    const int khalf = lane >> 4;
    const int z = blockIdx.z;
    const int b = z / H, h = z % H;
    const int mBase = blockIdx.y << 5;
    const int nBase = blockIdx.x << 6;

    v8f acc[8];
#pragma unroll
    for (int t = 0; t < 8; ++t)
#pragma unroll
        for (int r = 0; r < 8; ++r) acc[t][r] = 0.0f;

    // Uniform bases (fold b,h scalar terms into the pointer)
    const bf16* Qb = Q + ((size_t)b * S) * E + h * DH;
    const bf16* Kb = Kmat + ((size_t)b * S) * E + h * DH;

    const unsigned kh16  = (unsigned)(khalf * 16);
    const unsigned qoff0 = (unsigned)(mBase + lrow) * (unsigned)E;
    const unsigned qoff1 = qoff0 + 16u * (unsigned)E;
    const unsigned koff0 = (unsigned)(nBase +  0 + lrow) * (unsigned)E + kh16;
    const unsigned koff1 = (unsigned)(nBase + 16 + lrow) * (unsigned)E + kh16;
    const unsigned koff2 = (unsigned)(nBase + 32 + lrow) * (unsigned)E + kh16;
    const unsigned koff3 = (unsigned)(nBase + 48 + lrow) * (unsigned)E + kh16;

#pragma unroll
    for (unsigned kk = 0; kk < 64; kk += 32) {
        v16bf a0 = load_a_frag(Qb, qoff0 + kk, khalf, 0xFFFFFFFFu);
        v16bf a1 = load_a_frag(Qb, qoff1 + kk, khalf, 0xFFFFFFFFu);
        v16bf b0 = load_b_ct(Kb, koff0 + kk), b1 = load_b_ct(Kb, koff1 + kk);
        v16bf b2 = load_b_ct(Kb, koff2 + kk), b3 = load_b_ct(Kb, koff3 + kk);
        wmma8(acc, a0, a1, b0, b1, b2, b3);
    }

    float* srow = scores + (size_t)z * S * S;
#pragma unroll
    for (int half = 0; half < 2; ++half) {
#pragma unroll
        for (int t = 0; t < 4; ++t) {
            unsigned col = (unsigned)(nBase + t * 16 + lrow);
#pragma unroll
            for (int r = 0; r < 8; ++r) {
                unsigned row = (unsigned)(mBase + half * 16 + r + khalf * 8);
                srow[row * (unsigned)S + col] = acc[half * 4 + t][r] * 0.125f;  // 1/sqrt(64)
            }
        }
    }
}

// ---------------------------------------------------------------------------
// Row softmax in place. grid = B*H*S, 256 threads
// ---------------------------------------------------------------------------
__global__ void softmax_kernel(float* __restrict__ p, int N) {
    __shared__ float red[256];
    const int tid = threadIdx.x;
    float* pr = p + (size_t)blockIdx.x * N;

    float mx = -3.0e38f;
    for (int i = tid; i < N; i += 256) mx = fmaxf(mx, pr[i]);
    red[tid] = mx; __syncthreads();
    for (int off = 128; off > 0; off >>= 1) {
        if (tid < off) red[tid] = fmaxf(red[tid], red[tid + off]);
        __syncthreads();
    }
    float m = red[0]; __syncthreads();

    float s = 0.0f;
    for (int i = tid; i < N; i += 256) { float e = __expf(pr[i] - m); pr[i] = e; s += e; }
    red[tid] = s; __syncthreads();
    for (int off = 128; off > 0; off >>= 1) {
        if (tid < off) red[tid] += red[tid + off];
        __syncthreads();
    }
    float inv = 1.0f / red[0]; __syncthreads();
    for (int i = tid; i < N; i += 256) pr[i] *= inv;
}

// ---------------------------------------------------------------------------
// ctx[b,i,h,:] = sum_j P[b,h,i,j] * V[b,j,h,:]
// A = f32 probs (cvt->bf16 on the fly); B = Vt [B,H,DH,S] (contiguous-pair loads).
// grid = (S/32, B*H), 32 threads, 32x64 tile (N = DH = 64).
// ---------------------------------------------------------------------------
__global__ void __launch_bounds__(32, 1)
attn_apply_kernel(const float* __restrict__ P, const bf16* __restrict__ Vt,
                  bf16* __restrict__ ctx, int S, int E, int H, int DH) {
    const int lane  = threadIdx.x & 31;
    const int lrow  = lane & 15;
    const int khalf = lane >> 4;
    const int z = blockIdx.y;
    const int b = z / H, h = z % H;
    const int mBase = blockIdx.x << 5;

    v8f acc[8];
#pragma unroll
    for (int t = 0; t < 8; ++t)
#pragma unroll
        for (int r = 0; r < 8; ++r) acc[t][r] = 0.0f;

    const float* Pb = P + (size_t)z * S * S;            // uniform base
    const bf16*  Vb = Vt + (size_t)z * DH * S;          // uniform base

    const unsigned kh16  = (unsigned)(khalf * 16);
    const unsigned poff0 = (unsigned)(mBase + lrow) * (unsigned)S;
    const unsigned poff1 = poff0 + 16u * (unsigned)S;
    const unsigned voff0 = (unsigned)( 0 + lrow) * (unsigned)S + kh16;
    const unsigned voff1 = (unsigned)(16 + lrow) * (unsigned)S + kh16;
    const unsigned voff2 = (unsigned)(32 + lrow) * (unsigned)S + kh16;
    const unsigned voff3 = (unsigned)(48 + lrow) * (unsigned)S + kh16;

    for (unsigned kk = 0; kk < (unsigned)S; kk += 32) {
        union { v16bf v; unsigned u[8]; } a0, a1;
#pragma unroll
        for (int i = 0; i < 8; ++i) {
            unsigned k = kk + (unsigned)(2 * i + khalf * 16);
            a0.u[i] = packf2(Pb[poff0 + k], Pb[poff0 + k + 1]);
            a1.u[i] = packf2(Pb[poff1 + k], Pb[poff1 + k + 1]);
        }
        v16bf b0 = load_b_ct(Vb, voff0 + kk), b1 = load_b_ct(Vb, voff1 + kk);
        v16bf b2 = load_b_ct(Vb, voff2 + kk), b3 = load_b_ct(Vb, voff3 + kk);
        wmma8(acc, a0.v, a1.v, b0, b1, b2, b3);
    }

    bf16* cb = ctx + ((size_t)b * S) * E + h * DH;      // uniform base
#pragma unroll
    for (int half = 0; half < 2; ++half) {
#pragma unroll
        for (int t = 0; t < 4; ++t) {
            unsigned col = (unsigned)(t * 16 + lrow);
#pragma unroll
            for (int r = 0; r < 8; ++r) {
                unsigned row = (unsigned)(mBase + half * 16 + r + khalf * 8);
                cb[row * (unsigned)E + col] = (bf16)acc[half * 4 + t][r];
            }
        }
    }
}

// ---------------------------------------------------------------------------
// Tap-summed causal conv as GEMM: y[m,o] = sum_tap sum_c x[b, t+tap-2, c] * wr[tap][o][c]
// grid = (N/64, M/32), 32 threads; branchless per-lane zero mask for t' < 0.
// Register double-buffered K loop per tap.
// ---------------------------------------------------------------------------
__global__ void __launch_bounds__(32, 1)
conv_gemm_kernel(const bf16* __restrict__ A, const bf16* __restrict__ Wr,
                 const float* __restrict__ bias, const float* __restrict__ residual,
                 bf16* __restrict__ outb, float* __restrict__ outf,
                 int S, int Cin, int N, int relu) {
    const int lane  = threadIdx.x & 31;
    const int lrow  = lane & 15;
    const int khalf = lane >> 4;
    const int mBase = blockIdx.y << 5;
    const int nBase = blockIdx.x << 6;

    const int mrow0 = mBase + lrow;
    const int mrow1 = mrow0 + 16;
    const int b0r = mrow0 / S, t0r = mrow0 % S;
    const int b1r = mrow1 / S, t1r = mrow1 % S;

    v8f acc[8];
#pragma unroll
    for (int t = 0; t < 8; ++t)
#pragma unroll
        for (int r = 0; r < 8; ++r) acc[t][r] = 0.0f;

    const unsigned kh16 = (unsigned)(khalf * 16);

#pragma unroll
    for (int tap = 0; tap < 3; ++tap) {
        int tp0 = t0r + tap - 2;
        int tp1 = t1r + tap - 2;
        unsigned mask0 = (tp0 < 0) ? 0u : 0xFFFFFFFFu;
        unsigned mask1 = (tp1 < 0) ? 0u : 0xFFFFFFFFu;
        unsigned aoff0 = (unsigned)(b0r * S + ((tp0 < 0) ? 0 : tp0)) * (unsigned)Cin;
        unsigned aoff1 = (unsigned)(b1r * S + ((tp1 < 0) ? 0 : tp1)) * (unsigned)Cin;
        unsigned wbase = (unsigned)tap * (unsigned)N * (unsigned)Cin;
        unsigned woff0 = wbase + (unsigned)(nBase +  0 + lrow) * (unsigned)Cin + kh16;
        unsigned woff1 = wbase + (unsigned)(nBase + 16 + lrow) * (unsigned)Cin + kh16;
        unsigned woff2 = wbase + (unsigned)(nBase + 32 + lrow) * (unsigned)Cin + kh16;
        unsigned woff3 = wbase + (unsigned)(nBase + 48 + lrow) * (unsigned)Cin + kh16;

        v16bf a0 = load_a_frag(A, aoff0, khalf, mask0);
        v16bf a1 = load_a_frag(A, aoff1, khalf, mask1);
        v16bf b0 = load_b_ct(Wr, woff0), b1 = load_b_ct(Wr, woff1);
        v16bf b2 = load_b_ct(Wr, woff2), b3 = load_b_ct(Wr, woff3);

        for (unsigned kk = 32; kk < (unsigned)Cin; kk += 32) {
            v16bf a0n = load_a_frag(A, aoff0 + kk, khalf, mask0);
            v16bf a1n = load_a_frag(A, aoff1 + kk, khalf, mask1);
            v16bf b0n = load_b_ct(Wr, woff0 + kk), b1n = load_b_ct(Wr, woff1 + kk);
            v16bf b2n = load_b_ct(Wr, woff2 + kk), b3n = load_b_ct(Wr, woff3 + kk);
            wmma8(acc, a0, a1, b0, b1, b2, b3);
            a0 = a0n; a1 = a1n; b0 = b0n; b1 = b1n; b2 = b2n; b3 = b3n;
        }
        wmma8(acc, a0, a1, b0, b1, b2, b3);
    }

#pragma unroll
    for (int half = 0; half < 2; ++half) {
#pragma unroll
        for (int u = 0; u < 4; ++u) {
            int col = nBase + u * 16 + lrow;
            float bvv = bias[col];
#pragma unroll
            for (int r = 0; r < 8; ++r) {
                unsigned row = (unsigned)(mBase + half * 16 + r + khalf * 8);
                unsigned idx = row * (unsigned)N + (unsigned)col;
                float v = acc[half * 4 + u][r] + bvv;
                if (relu) v = fmaxf(v, 0.0f);
                if (residual) v += residual[idx];
                if (outf) outf[idx] = v;
                if (outb) outb[idx] = (bf16)v;
            }
        }
    }
}

// ---------------------------------------------------------------------------
// LayerNorm over last dim (gamma=1, beta=0, eps=1e-5). grid = #rows, 256 threads.
// ---------------------------------------------------------------------------
__global__ void layernorm_kernel(const float* __restrict__ x, float* __restrict__ outf,
                                 bf16* __restrict__ outb, int N) {
    __shared__ float red[256];
    const int tid = threadIdx.x;
    const float* xr = x + (size_t)blockIdx.x * N;

    float s = 0.0f;
    for (int i = tid; i < N; i += 256) s += xr[i];
    red[tid] = s; __syncthreads();
    for (int off = 128; off > 0; off >>= 1) {
        if (tid < off) red[tid] += red[tid + off];
        __syncthreads();
    }
    float mean = red[0] / (float)N; __syncthreads();

    float vs = 0.0f;
    for (int i = tid; i < N; i += 256) { float d = xr[i] - mean; vs += d * d; }
    red[tid] = vs; __syncthreads();
    for (int off = 128; off > 0; off >>= 1) {
        if (tid < off) red[tid] += red[tid + off];
        __syncthreads();
    }
    float inv = rsqrtf(red[0] / (float)N + 1e-5f);

    for (int i = tid; i < N; i += 256) {
        float v = (xr[i] - mean) * inv;
        outf[(size_t)blockIdx.x * N + i] = v;
        if (outb) outb[(size_t)blockIdx.x * N + i] = (bf16)v;
    }
}

// ---------------------------------------------------------------------------
// Host orchestration
// ---------------------------------------------------------------------------
extern "C" void kernel_launch(void* const* d_in, const int* in_sizes, int n_in,
                              void* d_out, int out_size, void* d_ws, size_t ws_size,
                              hipStream_t stream) {
    constexpr int Bc = 8, Sc = 1024, Ec = 1024, Hc = 16, DHc = 64, DFFc = 2048;
    constexpr int Mc = Bc * Sc;  // 8192 token rows

    const float* word = (const float*)d_in[0];
    const float* Wq = (const float*)d_in[1];  const float* bq = (const float*)d_in[2];
    const float* Wk = (const float*)d_in[3];  const float* bk = (const float*)d_in[4];
    const float* Wv = (const float*)d_in[5];  const float* bv = (const float*)d_in[6];
    const float* Wo = (const float*)d_in[7];  const float* bo = (const float*)d_in[8];
    const float* c1w = (const float*)d_in[9];  const float* c1b = (const float*)d_in[10];
    const float* c2w = (const float*)d_in[11]; const float* c2b = (const float*)d_in[12];

    float* out_final = (float*)d_out;                           // [B,S,E]
    float* attnp     = (float*)d_out + (size_t)Mc * Ec;         // [B,H,S,S]

    // Workspace carve-out
    size_t off = 0;
    char* wsc = (char*)d_ws;
    auto alloc = [&](size_t bytes) -> void* {
        void* p = (void*)(wsc + off);
        off += (bytes + 255) & ~(size_t)255;
        return p;
    };
    bf16*  word_b = (bf16*)alloc((size_t)Mc * Ec * 2);
    bf16*  Wqb    = (bf16*)alloc((size_t)Ec * Ec * 2);
    bf16*  Wkb    = (bf16*)alloc((size_t)Ec * Ec * 2);
    bf16*  Wvb    = (bf16*)alloc((size_t)Ec * Ec * 2);
    bf16*  Wob    = (bf16*)alloc((size_t)Ec * Ec * 2);
    bf16*  w1r    = (bf16*)alloc((size_t)3 * DFFc * Ec * 2);
    bf16*  w2r    = (bf16*)alloc((size_t)3 * Ec * DFFc * 2);
    bf16*  Qb     = (bf16*)alloc((size_t)Mc * Ec * 2);
    bf16*  Kb     = (bf16*)alloc((size_t)Mc * Ec * 2);
    bf16*  Vb     = (bf16*)alloc((size_t)Mc * Ec * 2);
    bf16*  Vtb    = (bf16*)alloc((size_t)Mc * Ec * 2);
    bf16*  ctx    = (bf16*)alloc((size_t)Mc * Ec * 2);
    float* tmpf   = (float*)alloc((size_t)Mc * Ec * 4);
    float* x1f    = (float*)alloc((size_t)Mc * Ec * 4);
    bf16*  x1b    = (bf16*)alloc((size_t)Mc * Ec * 2);
    bf16*  hb     = (bf16*)alloc((size_t)Mc * DFFc * 2);

    // 1) Convert fp32 -> bf16 / repack conv weights
    {
        size_t n = (size_t)Mc * Ec;
        cvt_f32_bf16_kernel<<<dim3((unsigned)((n + 255) / 256)), 256, 0, stream>>>(word, word_b, n);
        size_t nw = (size_t)Ec * Ec;
        unsigned gw = (unsigned)((nw + 255) / 256);
        cvt_f32_bf16_kernel<<<gw, 256, 0, stream>>>(Wq, Wqb, nw);
        cvt_f32_bf16_kernel<<<gw, 256, 0, stream>>>(Wk, Wkb, nw);
        cvt_f32_bf16_kernel<<<gw, 256, 0, stream>>>(Wv, Wvb, nw);
        cvt_f32_bf16_kernel<<<gw, 256, 0, stream>>>(Wo, Wob, nw);
        size_t nc = (size_t)DFFc * Ec * 3;
        repack_conv_kernel<<<(unsigned)((nc + 255) / 256), 256, 0, stream>>>(c1w, w1r, DFFc, Ec);
        repack_conv_kernel<<<(unsigned)((nc + 255) / 256), 256, 0, stream>>>(c2w, w2r, Ec, DFFc);
    }

    // 2) QKV projections (WMMA GEMMs)
    {
        dim3 grid(Ec / 64, Mc / 32);
        gemm_bias_kernel<<<grid, 32, 0, stream>>>(word_b, Wqb, bq, nullptr, Qb, nullptr, Mc, Ec, Ec);
        gemm_bias_kernel<<<grid, 32, 0, stream>>>(word_b, Wkb, bk, nullptr, Kb, nullptr, Mc, Ec, Ec);
        gemm_bias_kernel<<<grid, 32, 0, stream>>>(word_b, Wvb, bv, nullptr, Vb, nullptr, Mc, Ec, Ec);
    }

    // 3) V transpose for contiguous B-frags in P@V
    {
        size_t n = (size_t)Mc * Ec;
        transpose_v_kernel<<<(unsigned)((n + 255) / 256), 256, 0, stream>>>(Vb, Vtb, Sc, Ec, Hc, DHc);
    }

    // 4) Attention: scores -> softmax (into d_out attn region) -> P @ V
    attn_scores_kernel<<<dim3(Sc / 64, Sc / 32, Bc * Hc), 32, 0, stream>>>(Qb, Kb, attnp, Sc, Ec, Hc, DHc);
    softmax_kernel<<<(unsigned)(Bc * Hc * Sc), 256, 0, stream>>>(attnp, Sc);
    attn_apply_kernel<<<dim3(Sc / 32, Bc * Hc), 32, 0, stream>>>(attnp, Vtb, ctx, Sc, Ec, Hc, DHc);

    // 5) Output projection + residual, then LayerNorm -> x1 (f32 + bf16)
    gemm_bias_kernel<<<dim3(Ec / 64, Mc / 32), 32, 0, stream>>>(ctx, Wob, bo, word, nullptr, tmpf, Mc, Ec, Ec);
    layernorm_kernel<<<(unsigned)Mc, 256, 0, stream>>>(tmpf, x1f, x1b, Ec);

    // 6) Causal conv FFN as tap-summed WMMA GEMMs
    conv_gemm_kernel<<<dim3(DFFc / 64, Mc / 32), 32, 0, stream>>>(x1b, w1r, c1b, nullptr, hb, nullptr,
                                                                  Sc, Ec, DFFc, 1);
    conv_gemm_kernel<<<dim3(Ec / 64, Mc / 32), 32, 0, stream>>>(hb, w2r, c2b, x1f, nullptr, tmpf,
                                                                Sc, DFFc, Ec, 0);

    // 7) Final LayerNorm -> d_out "out" region
    layernorm_kernel<<<(unsigned)Mc, 256, 0, stream>>>(tmpf, out_final, nullptr, Ec);

    (void)in_sizes; (void)n_in; (void)out_size; (void)ws_size;
}
